// MultiModeCantorMoELayer_29626684407955
// MI455X (gfx1250) — compile-verified, matrix operand order checked by reading, not run on hardware
//
#include <hip/hip_runtime.h>
#include <hip/hip_bf16.h>
#include <math.h>

// Problem constants (from reference)
#define E_   8
#define DF_  512
#define DE_  128
#define H_   8
#define S_   64
#define GH_  16
#define HD_  16
#define B_   2
#define N_   1024
#define BN_  (B_*N_)

typedef _Float16 v16h __attribute__((ext_vector_type(16)));
typedef _Float16 v8h  __attribute__((ext_vector_type(8)));
typedef float    v8f  __attribute__((ext_vector_type(8)));

// Load a 16x32 (rows x K) f16 fragment from a row-major matrix with leading
// dim `ld` (halves). Matches CDNA5 16-bit A/B operand layout:
//   lane = row (lane&15), ks = (lane>>4)*8
//   elements 0..7  = K = ks..ks+7
//   elements 8..15 = K = 16+ks..16+ks+7
__device__ __forceinline__ v16h ldfrag(const _Float16* p, int ld) {
  int lane = threadIdx.x & 31;
  const _Float16* row = p + (lane & 15) * ld + ((lane >> 4) << 3);
  v8h lo = *(const v8h*)(row);
  v8h hi = *(const v8h*)(row + 16);
  v16h r;
#pragma unroll
  for (int i = 0; i < 8; ++i) { r[i] = lo[i]; r[i + 8] = hi[i]; }
  return r;
}

__device__ __forceinline__ v8f wmma16(v16h a, v16h b, v8f c) {
  // D = A(16x32) * B(32x16) + C, f32 accumulate
  return __builtin_amdgcn_wmma_f32_16x16x32_f16(false, a, false, b, (short)0, c,
                                                false, false);
}

// Async copy of 16 bytes global -> LDS (CDNA5 GLOBAL_LOAD_ASYNC_TO_LDS_B128,
// tracked by ASYNCcnt; no VGPR data round-trip).
__device__ __forceinline__ void async_copy_b128(const void* gsrc, void* lds_dst) {
  unsigned int dst = (unsigned int)(unsigned long long)lds_dst;  // low 32 = LDS offset
  unsigned long long src = (unsigned long long)gsrc;
  asm volatile("global_load_async_to_lds_b128 %0, %1, off"
               :: "v"(dst), "v"(src) : "memory");
}

// ---------------------------------------------------------------------------
// Kernel 0: weight conversion to f16 + per-token expert id from fingerprints
// ---------------------------------------------------------------------------
__global__ void prep_kernel(const float* __restrict__ Wq, const float* __restrict__ Wk,
                            const float* __restrict__ Wv, const float* __restrict__ Wf,
                            const float* __restrict__ fp,
                            _Float16* __restrict__ Wq16, _Float16* __restrict__ Wk16,
                            _Float16* __restrict__ Wv16, _Float16* __restrict__ Wf16,
                            int* __restrict__ eid) {
  int idx = blockIdx.x * blockDim.x + threadIdx.x;
  if (idx < E_ * DE_ * S_) {            // 65536 == DF_*DE_ as well
    Wq16[idx] = (_Float16)Wq[idx];
    Wk16[idx] = (_Float16)Wk[idx];
    Wv16[idx] = (_Float16)Wv[idx];
    Wf16[idx] = (_Float16)Wf[idx];
  }
  if (idx < N_) {
    float f = fp[idx];
    int e = (int)floorf(f * (float)E_);
    e = e < 0 ? 0 : (e > E_ - 1 ? E_ - 1 : e);
    eid[idx] = e;
  }
}

// ---------------------------------------------------------------------------
// Kernel 1: LayerNorm + per-expert MLP gate; emits gated slices fhat (f16)
// One wave per token; 8 waves per block.
// ---------------------------------------------------------------------------
__global__ void ln_gate_kernel(const float* __restrict__ x, const float* __restrict__ ln_w,
                               const float* __restrict__ ln_b, const float* __restrict__ alpha,
                               const float* __restrict__ Wg1, const float* __restrict__ bg1,
                               const float* __restrict__ Wg2, const float* __restrict__ bg2,
                               _Float16* __restrict__ fhat) {
  __shared__ float xns[8][DF_];
  int wave = threadIdx.x >> 5;
  int lane = threadIdx.x & 31;
  int t = blockIdx.x * 8 + wave;            // token id in [0, BN)
  const float* xr = x + (size_t)t * DF_;

  float sum = 0.f, sq = 0.f;
  float vals[16];
#pragma unroll
  for (int i = 0; i < 16; ++i) {
    float v = xr[lane * 16 + i];
    vals[i] = v; sum += v; sq += v * v;
  }
#pragma unroll
  for (int d = 1; d < 32; d <<= 1) {
    sum += __shfl_xor(sum, d, 32);
    sq  += __shfl_xor(sq, d, 32);
  }
  float mu = sum * (1.f / DF_);
  float var = sq * (1.f / DF_) - mu * mu;
  float rs = rsqrtf(var + 1e-5f);
#pragma unroll
  for (int i = 0; i < 16; ++i) {
    int j = lane * 16 + i;
    xns[wave][j] = (vals[i] - mu) * rs * ln_w[j] + ln_b[j];
  }
  __syncthreads();

  for (int e = 0; e < E_; ++e) {
    float part = 0.f;
    if (lane < GH_) {
      float acc = bg1[e * GH_ + lane];
      const float* wg = Wg1 + (size_t)(e * GH_ + lane) * S_;
      const float* fx = &xns[wave][e * S_];
#pragma unroll 8
      for (int s = 0; s < S_; ++s) acc += wg[s] * fx[s];
      float g1 = 0.5f * acc * (1.f + erff(acc * 0.70710678118f));  // exact gelu
      part = g1 * Wg2[e * GH_ + lane];
    }
#pragma unroll
    for (int d = 1; d < 16; d <<= 1) part += __shfl_xor(part, d, 32);
    float g2 = __shfl(part, 0, 32) + bg2[e];
    float sig = 1.f / (1.f + __expf(-g2));
    float aw  = 1.f / (1.f + __expf(-alpha[e]));
    float scale = sig * aw + (1.f - aw);
    _Float16* fo = fhat + ((size_t)e * BN_ + t) * S_;
#pragma unroll
    for (int j = 0; j < 2; ++j) {
      int c = 2 * lane + j;
      fo[c] = (_Float16)(xns[wave][e * S_ + c] * scale);
    }
  }
}

// ---------------------------------------------------------------------------
// Kernel 2: QKV projections via WMMA. Block = 8 waves, wave w handles head w
// (a 16-wide slice of DE). Q is pre-scaled by 1/(sqrt(HD)*|temp|).
// Q,K stored zero-padded to 32 along HD; V stored transposed (hd-major).
// ---------------------------------------------------------------------------
__global__ void qkv_kernel(const _Float16* __restrict__ fhat,
                           const _Float16* __restrict__ Wq16,
                           const _Float16* __restrict__ Wk16,
                           const _Float16* __restrict__ Wv16,
                           const float* __restrict__ temp,
                           _Float16* __restrict__ Qp, _Float16* __restrict__ Kp,
                           _Float16* __restrict__ Vt) {
  int e = blockIdx.x;
  int tile = blockIdx.y;                // 16-token tile, [0,128)
  int h = threadIdx.x >> 5;             // wave id == head id
  int lane = threadIdx.x & 31;
  const _Float16* fa = fhat + ((size_t)e * BN_ + tile * 16) * S_;
  v16h a0 = ldfrag(fa, S_);
  v16h a1 = ldfrag(fa + 32, S_);
  float qscale = 1.f / (4.f * fabsf(temp[0]));  // sqrt(HD)=4
  int hd = lane & 15;
  int hlf = lane >> 4;
  for (int tsel = 0; tsel < 3; ++tsel) {
    const _Float16* Wsrc = (tsel == 0 ? Wq16 : (tsel == 1 ? Wk16 : Wv16));
    const _Float16* W = Wsrc + (size_t)e * DE_ * S_ + (size_t)h * 16 * S_;
    v8f c = {};
    c = wmma16(a0, ldfrag(W, S_), c);
    c = wmma16(a1, ldfrag(W + 32, S_), c);
#pragma unroll
    for (int v = 0; v < 8; ++v) {
      int m = v + 8 * hlf;
      int tok = tile * 16 + m;
      int b = tok >> 10, n = tok & (N_ - 1);
      size_t hb = (size_t)(e * B_ + b) * H_ + h;
      if (tsel == 0) {
        _Float16* q = Qp + (hb * N_ + n) * 32;
        q[hd] = (_Float16)(c[v] * qscale);
        q[hd + 16] = (_Float16)0.f;     // zero pad (HD=16 -> K=32)
      } else if (tsel == 1) {
        _Float16* k = Kp + (hb * N_ + n) * 32;
        k[hd] = (_Float16)c[v];
        k[hd + 16] = (_Float16)0.f;
      } else {
        Vt[(hb * 16 + hd) * N_ + n] = (_Float16)c[v];
      }
    }
  }
}

// ---------------------------------------------------------------------------
// Kernel 3: flash attention per (e,b,h). 8 waves/block share the same K/V
// stream, so each 32-key chunk of K (2KB) and V (1KB) is staged into LDS once
// per block with async global->LDS b128 copies (ASYNCcnt), double-buffered to
// overlap the copy of chunk k+1 with WMMA+softmax on chunk k. Key-side bucket
// mask; online softmax; P goes through a per-wave LDS tile to convert the
// WMMA C-layout into an A-fragment layout.
// ---------------------------------------------------------------------------
__global__ void attn_kernel(const _Float16* __restrict__ Qp,
                            const _Float16* __restrict__ Kp,
                            const _Float16* __restrict__ Vt,
                            const int* __restrict__ eid,
                            _Float16* __restrict__ fused16) {
  __shared__ alignas(16) _Float16 Kbuf[2][32][32];   // [buf][key][hd-padded]
  __shared__ alignas(16) _Float16 Vbuf[2][16][32];   // [buf][hd][key]
  __shared__ alignas(16) _Float16 Pb[8][16][32];     // per-wave P staging
  int idx = blockIdx.x;                 // (e*B + b)*H + h
  int e = idx >> 4;
  int b = (idx >> 3) & 1;
  int h = idx & 7;
  int tid = threadIdx.x;
  int wave = tid >> 5;
  int lane = tid & 31;
  int q0 = (blockIdx.y * 8 + wave) * 16;
  size_t hb = (size_t)(e * B_ + b) * H_ + h;
  const _Float16* Qb = Qp + (hb * N_ + q0) * 32;
  const _Float16* Kb = Kp + hb * N_ * 32;
  const _Float16* Vb = Vt + hb * 16 * N_;
  v16h aq = ldfrag(Qb, 32);
  v8f acc = {};
  float mrow[8], lrow[8];
#pragma unroll
  for (int v = 0; v < 8; ++v) { mrow[v] = -1e30f; lrow[v] = 0.f; }
  int col = lane & 15;
  int hlf = lane >> 4;
  v8f zero = {};

  // Stage one 32-key chunk of K and V into LDS buffer `buf` (async).
  auto stage = [&](int kb, int buf) {
    if (tid < 128) {
      // K chunk: 32 rows x 32 halves contiguous = 2KB = 128 x 16B
      const _Float16* src = Kb + (size_t)kb * 32 * 32 + tid * 8;
      async_copy_b128(src, (char*)&Kbuf[buf][0][0] + tid * 16);
    } else if (tid < 192) {
      // V chunk: 16 rows (hd) x 32 halves, global row stride N_
      int j = tid - 128;            // 0..63
      int r = j >> 2;               // hd row 0..15
      int c = (j & 3) * 8;          // half offset 0,8,16,24
      const _Float16* src = Vb + (size_t)r * N_ + kb * 32 + c;
      async_copy_b128(src, &Vbuf[buf][r][c]);
    }
  };

  const int NKB = N_ / 32;
  stage(0, 0);
  for (int kb = 0; kb < NKB; ++kb) {
    int cur = kb & 1;
    int k0 = kb * 32;
    if (kb + 1 < NKB) {
      stage(kb + 1, cur ^ 1);
      asm volatile("s_wait_asynccnt 0x1" ::: "memory");  // chunk kb resident
    } else {
      asm volatile("s_wait_asynccnt 0x0" ::: "memory");
    }
    __syncthreads();

    v16h bk0 = ldfrag(&Kbuf[cur][0][0], 32);
    v16h bk1 = ldfrag(&Kbuf[cur][16][0], 32);
    v8f s0 = wmma16(aq, bk0, zero);
    v8f s1 = wmma16(aq, bk1, zero);
    float msk0 = (eid[k0 + col] == e) ? 0.f : -1e9f;
    float msk1 = (eid[k0 + 16 + col] == e) ? 0.f : -1e9f;
#pragma unroll
    for (int v = 0; v < 8; ++v) {
      float a0 = s0[v] + msk0;
      float a1 = s1[v] + msk1;
      float rm = fmaxf(a0, a1);
#pragma unroll
      for (int d = 1; d < 16; d <<= 1) rm = fmaxf(rm, __shfl_xor(rm, d, 32));
      float mn = fmaxf(mrow[v], rm);
      float sc = __expf(mrow[v] - mn);
      mrow[v] = mn;
      float p0 = __expf(a0 - mn);
      float p1 = __expf(a1 - mn);
      float ps = p0 + p1;
#pragma unroll
      for (int d = 1; d < 16; d <<= 1) ps += __shfl_xor(ps, d, 32);
      lrow[v] = lrow[v] * sc + ps;
      acc[v] *= sc;
      int m = v + 8 * hlf;
      Pb[wave][m][col]      = (_Float16)p0;  // C-layout -> row-major staging
      Pb[wave][m][col + 16] = (_Float16)p1;
    }
    __syncthreads();
    v16h ap = ldfrag(&Pb[wave][0][0], 32);   // re-load as A fragment
    v16h bv = ldfrag(&Vbuf[cur][0][0], 32);  // rows=hd, K=keys
    acc = wmma16(ap, bv, acc);
    __syncthreads();   // consumers done before buffers are overwritten
  }
#pragma unroll
  for (int v = 0; v < 8; ++v) {
    int m = v + 8 * hlf;
    int q = q0 + m;
    if (eid[q] == e)
      fused16[((size_t)b * N_ + q) * DE_ + h * 16 + col] =
          (_Float16)(acc[v] / lrow[v]);
  }
}

// ---------------------------------------------------------------------------
// Kernel 4: output projection fused(2048x128) @ Wf^T(128x512) + bf + residual
// ---------------------------------------------------------------------------
__global__ void proj_kernel(const _Float16* __restrict__ fused16,
                            const _Float16* __restrict__ Wf16,
                            const float* __restrict__ x, const float* __restrict__ bf,
                            float* __restrict__ out) {
  int tile = blockIdx.x;                      // 16-token tile
  int wave = threadIdx.x >> 5;
  int lane = threadIdx.x & 31;
  int d0 = (blockIdx.y * 8 + wave) * 16;      // output column tile
  int tok0 = tile * 16;
  v8f acc = {};
#pragma unroll
  for (int k0 = 0; k0 < DE_; k0 += 32) {
    v16h a  = ldfrag(fused16 + (size_t)tok0 * DE_ + k0, DE_);
    v16h bw = ldfrag(Wf16 + (size_t)d0 * DE_ + k0, DE_);
    acc = wmma16(a, bw, acc);
  }
  int nd = d0 + (lane & 15);
  float bias = bf[nd];
#pragma unroll
  for (int v = 0; v < 8; ++v) {
    int m = v + 8 * (lane >> 4);
    size_t off = (size_t)(tok0 + m) * DF_ + nd;
    out[off] = x[off] + acc[v] + bias;
  }
}

// ---------------------------------------------------------------------------
extern "C" void kernel_launch(void* const* d_in, const int* in_sizes, int n_in,
                              void* d_out, int out_size, void* d_ws, size_t ws_size,
                              hipStream_t stream) {
  const float* x    = (const float*)d_in[0];
  const float* fp   = (const float*)d_in[1];
  const float* ln_w = (const float*)d_in[2];
  const float* ln_b = (const float*)d_in[3];
  const float* alpha= (const float*)d_in[4];
  const float* Wg1  = (const float*)d_in[5];
  const float* bg1  = (const float*)d_in[6];
  const float* Wg2  = (const float*)d_in[7];
  const float* bg2  = (const float*)d_in[8];
  const float* Wq   = (const float*)d_in[9];
  const float* Wk   = (const float*)d_in[10];
  const float* Wv   = (const float*)d_in[11];
  const float* temp = (const float*)d_in[12];
  const float* Wf   = (const float*)d_in[13];
  const float* bf   = (const float*)d_in[14];
  float* out = (float*)d_out;

  char* ws = (char*)d_ws;
  size_t off = 0;
  auto alloc = [&](size_t bytes) {
    char* p = ws + off;
    off = (off + bytes + 255) & ~(size_t)255;
    return p;
  };
  _Float16* fhat   = (_Float16*)alloc((size_t)E_ * BN_ * S_ * 2);        // 2 MB
  _Float16* Wq16   = (_Float16*)alloc((size_t)E_ * DE_ * S_ * 2);
  _Float16* Wk16   = (_Float16*)alloc((size_t)E_ * DE_ * S_ * 2);
  _Float16* Wv16   = (_Float16*)alloc((size_t)E_ * DE_ * S_ * 2);
  _Float16* Wf16   = (_Float16*)alloc((size_t)DF_ * DE_ * 2);
  int*      eid    = (int*)     alloc((size_t)N_ * 4);
  _Float16* Qp     = (_Float16*)alloc((size_t)E_ * B_ * H_ * N_ * 32 * 2); // 8 MB
  _Float16* Kp     = (_Float16*)alloc((size_t)E_ * B_ * H_ * N_ * 32 * 2); // 8 MB
  _Float16* Vt     = (_Float16*)alloc((size_t)E_ * B_ * H_ * 16 * N_ * 2); // 4 MB
  _Float16* fused16= (_Float16*)alloc((size_t)BN_ * DE_ * 2);              // 0.5 MB
  (void)ws_size; (void)in_sizes; (void)n_in; (void)out_size;

  prep_kernel<<<dim3(256), dim3(256), 0, stream>>>(Wq, Wk, Wv, Wf, fp,
                                                   Wq16, Wk16, Wv16, Wf16, eid);
  ln_gate_kernel<<<dim3(BN_ / 8), dim3(256), 0, stream>>>(
      x, ln_w, ln_b, alpha, Wg1, bg1, Wg2, bg2, fhat);
  qkv_kernel<<<dim3(E_, BN_ / 16), dim3(256), 0, stream>>>(
      fhat, Wq16, Wk16, Wv16, temp, Qp, Kp, Vt);
  attn_kernel<<<dim3(E_ * B_ * H_, N_ / 128), dim3(256), 0, stream>>>(
      Qp, Kp, Vt, eid, fused16);
  proj_kernel<<<dim3(BN_ / 16, DF_ / 128), dim3(256), 0, stream>>>(
      fused16, Wf16, x, bf, out);
}
